// SpatialWeightedAttention_90486370992250
// MI455X (gfx1250) — matrix operand order repaired and checked by reading, not compiled, not run on hardware
//
#include <hip/hip_runtime.h>
#include <hip/hip_bf16.h>

#define B_  4
#define N_  1024
#define D_  1024
#define H_  16
#define HD_ 64

typedef __attribute__((ext_vector_type(16))) _Float16 v16h;
typedef __attribute__((ext_vector_type(8)))  _Float16 v8h;
typedef __attribute__((ext_vector_type(8)))  float    v8f;
typedef __attribute__((ext_vector_type(4)))  int      v4i;

// ---- gfx1250 async copy to LDS (ASYNCcnt path), with safe fallback ----
#if defined(__HIP_DEVICE_COMPILE__) && \
    __has_builtin(__builtin_amdgcn_global_load_async_to_lds_b128) && \
    __has_builtin(__builtin_amdgcn_s_wait_asynccnt)
#define USE_ASYNC_LDS 1
#else
#define USE_ASYNC_LDS 0
#endif

#if USE_ASYNC_LDS
typedef __attribute__((address_space(1))) v4i* g_v4i_ptr;   // global v4i*
typedef __attribute__((address_space(3))) v4i* l_v4i_ptr;   // LDS v4i*
#endif

__device__ __forceinline__ void stage16B(const _Float16* gsrc, _Float16* ldst) {
#if USE_ASYNC_LDS
    __builtin_amdgcn_global_load_async_to_lds_b128((g_v4i_ptr)gsrc, (l_v4i_ptr)ldst, 0, 0);
#else
    *(v8h*)ldst = *(const v8h*)gsrc;
#endif
}
#if USE_ASYNC_LDS
#define WAIT_ASYNC(n) __builtin_amdgcn_s_wait_asynccnt(n)
#else
#define WAIT_ASYNC(n) do {} while (0)
#endif

// ---- WMMA fragment loaders (gfx1250 wave32 layouts, 16-bit operands) ----
// A (16xK=32): lanes 0-15 row=lane, K {s..s+7, 16+s..} with s=0; lanes 16-31 s=8
__device__ __forceinline__ v16h load_a_frag(const _Float16* base, int ld, int lane) {
    int r = lane & 15;
    int s = (lane >> 4) << 3;
    const _Float16* p = base + (size_t)r * ld + s;
    v8h lo = *(const v8h*)(p);
    v8h hi = *(const v8h*)(p + 16);
    return __builtin_shufflevector(lo, hi, 0,1,2,3,4,5,6,7,8,9,10,11,12,13,14,15);
}

// B (K=32 x N=16): lane holds column n=l&15, K {s..s+7, 16+s..} with s=(l>>4)*8; source stored [n][k], stride ld
__device__ __forceinline__ v16h load_b_frag(const _Float16* base, int ld, int lane) {
    int c = lane & 15;
    int s = (lane >> 4) << 3;
    const _Float16* p = base + (size_t)c * ld + s;
    v8h lo = *(const v8h*)(p);
    v8h hi = *(const v8h*)(p + 16);
    return __builtin_shufflevector(lo, hi, 0,1,2,3,4,5,6,7,8,9,10,11,12,13,14,15);
}

// LDS B tile: 64 columns x 32 K halfs, padded row stride 40 halfs (80B, keeps 16B align)
#define BS_LD 40

// one k-step: 1 A fragment, 4 B fragments from LDS, 4 WMMAs (in-place accumulate)
__device__ __forceinline__ void gemm_step(const _Float16* arow, int k0,
                                          const _Float16* bbase, int lane, v8f* acc) {
    __builtin_prefetch((const void*)(arow + k0 + 128), 0, 1);
    v16h a = load_a_frag(arow + k0, D_, lane);
#pragma unroll
    for (int nn = 0; nn < 4; ++nn) {
        v16h b = load_b_frag(bbase + (nn * 16) * BS_LD, BS_LD, lane);
        acc[nn] = __builtin_amdgcn_wmma_f32_16x16x32_f16(false, a, false, b, (short)0, acc[nn], false, false);
    }
}

// ---- Kernel 1: LayerNorm (f32 in, f16 out) ----
__global__ void ln_kernel(const float* __restrict__ x, const float* __restrict__ gamma,
                          const float* __restrict__ beta, _Float16* __restrict__ xn) {
    int tok = blockIdx.x;
    int tid = threadIdx.x;              // 256 threads
    const float* row = x + (size_t)tok * D_;
    float v[4], s = 0.f, ss = 0.f;
#pragma unroll
    for (int i = 0; i < 4; ++i) { v[i] = row[tid + 256 * i]; s += v[i]; ss += v[i] * v[i]; }
    __shared__ float red0[256], red1[256];
    red0[tid] = s; red1[tid] = ss;
    __syncthreads();
    for (int w = 128; w > 0; w >>= 1) {
        if (tid < w) { red0[tid] += red0[tid + w]; red1[tid] += red1[tid + w]; }
        __syncthreads();
    }
    float mu   = red0[0] * (1.0f / D_);
    float var  = red1[0] * (1.0f / D_) - mu * mu;
    float rstd = rsqrtf(var + 1e-5f);
#pragma unroll
    for (int i = 0; i < 4; ++i) {
        int c = tid + 256 * i;
        xn[(size_t)tok * D_ + c] = (_Float16)(((v[i] - mu) * rstd) * gamma[c] + beta[c]);
    }
}

// ---- Kernel 2: convert + transpose weights to f16 [col][k] ----
__global__ void cvt_kernel(const float* __restrict__ wqkv, const float* __restrict__ wproj,
                           _Float16* __restrict__ wqkvT, _Float16* __restrict__ wprojT) {
    int t = blockIdx.x * blockDim.x + threadIdx.x;
    const int total1 = 3 * D_ * D_;
    if (t < total1) {
        int c = t / D_, k = t - c * D_;
        wqkvT[t] = (_Float16)wqkv[(size_t)k * (3 * D_) + c];
    } else {
        int t2 = t - total1;
        if (t2 < D_ * D_) {
            int c = t2 / D_, k = t2 - c * D_;
            wprojT[t2] = (_Float16)wproj[(size_t)k * D_ + c];
        }
    }
}

// ---- Kernel 3: QKV GEMM (128x64 block tile, 8 waves, async-LDS staged B) ----
__global__ void qkv_gemm_kernel(const _Float16* __restrict__ xn, const _Float16* __restrict__ wqkvT,
                                _Float16* __restrict__ qb, _Float16* __restrict__ kb,
                                _Float16* __restrict__ vT) {
    __shared__ __align__(16) _Float16 Bs[2][64 * BS_LD];
    int tid  = threadIdx.x;             // 256
    int lane = tid & 31, w = tid >> 5;  // 8 waves
    int tn0 = blockIdx.x * 64;
    int tm  = blockIdx.y * 128 + w * 16;

    int sc = tid >> 2;                  // staged column 0..63
    int sk = (tid & 3) * 8;             // k offset 0/8/16/24
    const _Float16* gB = wqkvT + (size_t)(tn0 + sc) * D_ + sk;
    _Float16* lB0 = &Bs[0][sc * BS_LD + sk];
    _Float16* lB1 = &Bs[1][sc * BS_LD + sk];

    const _Float16* arow = xn + (size_t)tm * D_;
    v8f acc[4] = {};

    stage16B(gB,      lB0);             // k-step 0 -> buf0
    stage16B(gB + 32, lB1);             // k-step 1 -> buf1
    for (int kt2 = 0; kt2 < 30; kt2 += 2) {     // steady state: branch-free stages
        WAIT_ASYNC(1);
        __syncthreads();
        gemm_step(arow, kt2 * 32, &Bs[0][0], lane, acc);
        __syncthreads();
        stage16B(gB + (kt2 + 2) * 32, lB0);
        WAIT_ASYNC(1);
        __syncthreads();
        gemm_step(arow, (kt2 + 1) * 32, &Bs[1][0], lane, acc);
        __syncthreads();
        stage16B(gB + (kt2 + 3) * 32, lB1);
    }
    // tail: k-steps 30 (buf0) and 31 (buf1); exactly two stages outstanding
    WAIT_ASYNC(1);
    __syncthreads();
    gemm_step(arow, 30 * 32, &Bs[0][0], lane, acc);
    WAIT_ASYNC(0);
    __syncthreads();
    gemm_step(arow, 31 * 32, &Bs[1][0], lane, acc);

    // scatter q/k [B,H,N,HD], v transposed [B,H,HD,N]
#pragma unroll
    for (int nn = 0; nn < 4; ++nn) {
        int col = tn0 + nn * 16 + (lane & 15);
        int which = col >> 10, rem = col & 1023;
        int h = rem >> 6, hd = rem & 63;
#pragma unroll
        for (int i = 0; i < 8; ++i) {
            int row = tm + i + ((lane >> 4) << 3);
            int b_ = row >> 10, tok = row & 1023;
            int bh = b_ * H_ + h;
            _Float16 val = (_Float16)acc[nn][i];
            if (which == 0)      qb[((size_t)bh * N_ + tok) * HD_ + hd] = val;
            else if (which == 1) kb[((size_t)bh * N_ + tok) * HD_ + hd] = val;
            else                 vT[((size_t)bh * HD_ + hd) * N_ + tok] = val;
        }
    }
}

// ---- Kernel 4: fused attention (flash-style, one wave per (b,h,16-query tile)) ----
__global__ void attn_kernel(const _Float16* __restrict__ qb, const _Float16* __restrict__ kb,
                            const _Float16* __restrict__ vT, const float* __restrict__ sim,
                            const float* __restrict__ swl_p, _Float16* __restrict__ ao) {
    int lane = threadIdx.x;
    int bh = blockIdx.x;
    int q0 = blockIdx.y * 16;
    int b_ = bh >> 4, h = bh & 15;

    float swl = *swl_p;
    float alpha = 1.0f / (1.0f + __expf(-swl));
    float c1 = (1.0f - alpha) * 0.125f; // scale = HD^-0.5

    const _Float16* qbase = qb + ((size_t)bh * N_ + q0) * HD_;
    v16h aq0 = load_a_frag(qbase, HD_, lane);
    v16h aq1 = load_a_frag(qbase + 32, HD_, lane);

    float m_run[8], l_run[8];
    v8f o[4] = {};
#pragma unroll
    for (int i = 0; i < 8; ++i) { m_run[i] = -3.0e38f; l_run[i] = 0.f; }

    __shared__ __align__(16) _Float16 Plds[16 * 32];
    int colid = lane & 15;
    int rbase = (lane >> 4) << 3;

    for (int j0 = 0; j0 < N_; j0 += 32) {
        const _Float16* kbase = kb + ((size_t)bh * N_ + j0) * HD_;
        v8f sfrag[2];
#pragma unroll
        for (int g = 0; g < 2; ++g) {
            v16h b0 = load_b_frag(kbase + (size_t)(16 * g) * HD_,      HD_, lane);
            v16h b1 = load_b_frag(kbase + (size_t)(16 * g) * HD_ + 32, HD_, lane);
            v8f c = {};
            c = __builtin_amdgcn_wmma_f32_16x16x32_f16(false, aq0, false, b0, (short)0, c, false, false);
            c = __builtin_amdgcn_wmma_f32_16x16x32_f16(false, aq1, false, b1, (short)0, c, false, false);
            sfrag[g] = c;
        }
#pragma unroll
        for (int g = 0; g < 2; ++g) {
            int scol = j0 + 16 * g + colid;
#pragma unroll
            for (int i = 0; i < 8; ++i) {
                int srow = q0 + rbase + i;
                sfrag[g][i] = c1 * sfrag[g][i] + alpha * sim[(size_t)srow * N_ + scol];
            }
        }
        float tmax[8];
#pragma unroll
        for (int i = 0; i < 8; ++i) tmax[i] = fmaxf(sfrag[0][i], sfrag[1][i]);
#pragma unroll
        for (int m = 1; m <= 8; m <<= 1)
#pragma unroll
            for (int i = 0; i < 8; ++i) tmax[i] = fmaxf(tmax[i], __shfl_xor(tmax[i], m, 32));
        float psum[8];
#pragma unroll
        for (int i = 0; i < 8; ++i) {
            float mn  = fmaxf(m_run[i], tmax[i]);
            float fac = __expf(m_run[i] - mn);
            m_run[i] = mn;
            l_run[i] *= fac;
#pragma unroll
            for (int t = 0; t < 4; ++t) o[t][i] *= fac;
            float p0 = __expf(sfrag[0][i] - mn);
            float p1 = __expf(sfrag[1][i] - mn);
            sfrag[0][i] = p0; sfrag[1][i] = p1;
            psum[i] = p0 + p1;
        }
#pragma unroll
        for (int m = 1; m <= 8; m <<= 1)
#pragma unroll
            for (int i = 0; i < 8; ++i) psum[i] += __shfl_xor(psum[i], m, 32);
#pragma unroll
        for (int i = 0; i < 8; ++i) l_run[i] += psum[i];
        __syncthreads();
#pragma unroll
        for (int g = 0; g < 2; ++g)
#pragma unroll
            for (int i = 0; i < 8; ++i)
                Plds[(rbase + i) * 32 + 16 * g + colid] = (_Float16)sfrag[g][i];
        __syncthreads();
        v16h ap = load_a_frag(Plds, 32, lane);
        const _Float16* vbase = vT + (size_t)bh * HD_ * N_ + j0;
#pragma unroll
        for (int t = 0; t < 4; ++t) {
            v16h bv = load_b_frag(vbase + (size_t)(16 * t) * N_, N_, lane);
            o[t] = __builtin_amdgcn_wmma_f32_16x16x32_f16(false, ap, false, bv, (short)0, o[t], false, false);
        }
    }
#pragma unroll
    for (int i = 0; i < 8; ++i) {
        float inv = 1.0f / l_run[i];
        int row = q0 + rbase + i;
        size_t base = ((size_t)b_ * N_ + row) * D_ + (size_t)h * HD_ + colid;
#pragma unroll
        for (int t = 0; t < 4; ++t)
            ao[base + 16 * t] = (_Float16)(o[t][i] * inv);
    }
}

// ---- Kernel 5: projection GEMM + bias + residual (128x64 block tile, async-LDS B) ----
__global__ void proj_kernel(const _Float16* __restrict__ ao, const _Float16* __restrict__ wprojT,
                            const float* __restrict__ x, const float* __restrict__ bproj,
                            float* __restrict__ out) {
    __shared__ __align__(16) _Float16 Bs[2][64 * BS_LD];
    int tid  = threadIdx.x;
    int lane = tid & 31, w = tid >> 5;
    int tn0 = blockIdx.x * 64;
    int tm  = blockIdx.y * 128 + w * 16;

    int sc = tid >> 2;
    int sk = (tid & 3) * 8;
    const _Float16* gB = wprojT + (size_t)(tn0 + sc) * D_ + sk;
    _Float16* lB0 = &Bs[0][sc * BS_LD + sk];
    _Float16* lB1 = &Bs[1][sc * BS_LD + sk];

    const _Float16* arow = ao + (size_t)tm * D_;
    v8f acc[4] = {};

    stage16B(gB,      lB0);
    stage16B(gB + 32, lB1);
    for (int kt2 = 0; kt2 < 30; kt2 += 2) {
        WAIT_ASYNC(1);
        __syncthreads();
        gemm_step(arow, kt2 * 32, &Bs[0][0], lane, acc);
        __syncthreads();
        stage16B(gB + (kt2 + 2) * 32, lB0);
        WAIT_ASYNC(1);
        __syncthreads();
        gemm_step(arow, (kt2 + 1) * 32, &Bs[1][0], lane, acc);
        __syncthreads();
        stage16B(gB + (kt2 + 3) * 32, lB1);
    }
    WAIT_ASYNC(1);
    __syncthreads();
    gemm_step(arow, 30 * 32, &Bs[0][0], lane, acc);
    WAIT_ASYNC(0);
    __syncthreads();
    gemm_step(arow, 31 * 32, &Bs[1][0], lane, acc);

#pragma unroll
    for (int nn = 0; nn < 4; ++nn) {
        int col = tn0 + nn * 16 + (lane & 15);
        float bias = bproj[col];
#pragma unroll
        for (int i = 0; i < 8; ++i) {
            int row = tm + i + ((lane >> 4) << 3);
            out[(size_t)row * D_ + col] = x[(size_t)row * D_ + col] + acc[nn][i] + bias;
        }
    }
}

extern "C" void kernel_launch(void* const* d_in, const int* in_sizes, int n_in,
                              void* d_out, int out_size, void* d_ws, size_t ws_size,
                              hipStream_t stream) {
    const float* x      = (const float*)d_in[0];
    const float* gamma  = (const float*)d_in[1];
    const float* beta   = (const float*)d_in[2];
    const float* w_qkv  = (const float*)d_in[3];
    const float* w_proj = (const float*)d_in[4];
    const float* b_proj = (const float*)d_in[5];
    const float* swl    = (const float*)d_in[6];
    const float* sim    = (const float*)d_in[7];
    float* out = (float*)d_out;

    char* ws = (char*)d_ws;
    _Float16* xn     = (_Float16*)(ws);                        //  8 MB [4096][1024]
    _Float16* wqkvT  = (_Float16*)(ws + ((size_t)8  << 20));   //  6 MB [3072][1024]
    _Float16* wprojT = (_Float16*)(ws + ((size_t)14 << 20));   //  2 MB [1024][1024]
    _Float16* qb     = (_Float16*)(ws + ((size_t)16 << 20));   //  8 MB [B,H,N,HD]
    _Float16* kb     = (_Float16*)(ws + ((size_t)24 << 20));   //  8 MB [B,H,N,HD]
    _Float16* vT     = (_Float16*)(ws + ((size_t)32 << 20));   //  8 MB [B,H,HD,N]
    _Float16* ao     = (_Float16*)(ws + ((size_t)40 << 20));   //  8 MB [B,N,D]

    ln_kernel<<<B_ * N_, 256, 0, stream>>>(x, gamma, beta, xn);
    cvt_kernel<<<(4 * D_ * D_ + 255) / 256, 256, 0, stream>>>(w_qkv, w_proj, wqkvT, wprojT);

    dim3 g1((3 * D_) / 64, (B_ * N_) / 128);
    qkv_gemm_kernel<<<g1, 256, 0, stream>>>(xn, wqkvT, qb, kb, vT);

    dim3 g2(B_ * H_, N_ / 16);
    attn_kernel<<<g2, 32, 0, stream>>>(qb, kb, vT, sim, swl, ao);

    dim3 g3(D_ / 64, (B_ * N_) / 128);
    proj_kernel<<<g3, 256, 0, stream>>>(ao, wprojT, x, b_proj, out);
}
